// EfficientAttention_17085379904421
// MI455X (gfx1250) — matrix operand order repaired and checked by reading, not compiled, not run on hardware
//
#include <hip/hip_runtime.h>

typedef __attribute__((ext_vector_type(16))) _Float16 v16h;
typedef __attribute__((ext_vector_type(8)))  float    v8f;

#define B_  4
#define C_  512
#define N_  4096
#define SCALE 0.044194173824159216f   // 512^-0.5
#define EPSV 1e-5f

static constexpr size_t SEG = (size_t)B_ * C_ * N_;   // 8388608 f16 elems

// ---------------- fragment-order swizzles (ISA 7.12.2, wave32) ----------------
// A-tile 16(row) x 32(k) f16 -> 512 contiguous f16: lane*16 + j
__device__ __forceinline__ int swA(int row, int kin) {
  return ((row + (((kin >> 3) & 1) << 4)) << 4) + ((kin & 7) | ((kin >> 1) & 8));
}
// B-tile 32(k) x 16(col) f16 -> 512 contiguous f16: lane*16 + j
__device__ __forceinline__ int swB(int k, int col) {
  return ((col + (((k >> 3) & 1) << 4)) << 4) + ((k & 7) | ((k >> 1) & 8));
}

__device__ __forceinline__ v8f wmma_f16(v16h a, v16h b, v8f c) {
  return __builtin_amdgcn_wmma_f32_16x16x32_f16(false, a, false, b, (short)0, c, false, false);
}

// global tensor indexers (fragment-tiled layouts)
// h  : [b][cChunk 16][nTile 256][512]   (B operand of QKV GEMM)
__device__ __forceinline__ size_t idx_h(int b, int c, int n) {
  return (((size_t)(b * 16 + (c >> 5)) * 256 + (n >> 4)) << 9) + swB(c & 31, n & 15);
}
// qT : [b][nTile 256][cChunk 16][512]   (A operand of S GEMM)
__device__ __forceinline__ size_t idx_q(int b, int n, int c) {
  return (((size_t)(b * 256 + (n >> 4)) * 16 + (c >> 5)) << 9) + swA(n & 15, c & 31);
}
// k  : [b][cChunk 16][mTile 256][512]   (B operand of S GEMM)
__device__ __forceinline__ size_t idx_k(int b, int c, int m) {
  return (((size_t)(b * 16 + (c >> 5)) * 256 + (m >> 4)) << 9) + swB(c & 31, m & 15);
}
// v  : [b][mChunk 128][cTile 32][512]   (B operand of P@V GEMM)
__device__ __forceinline__ size_t idx_v(int b, int m, int c) {
  return (((size_t)(b * 128 + (m >> 5)) * 32 + (c >> 4)) << 9) + swB(m & 31, c & 15);
}
// O  : [b][nTile 256][cChunk 16][512]   (A operand of proj GEMM)
__device__ __forceinline__ size_t idx_O(int b, int n, int c) {
  return (((size_t)(b * 256 + (n >> 4)) * 16 + (c >> 5)) << 9) + swA(n & 15, c & 31);
}
// w16: [cChunk 16][oTile 96][512]       (A operand of QKV GEMM)
__device__ __forceinline__ size_t idx_w(int o, int c) {
  return (((size_t)(c >> 5) * 96 + (o >> 4)) << 9) + swA(o & 15, c & 31);
}
// pw : [cChunk 16][oTile 32][512]       (B operand of proj GEMM)
__device__ __forceinline__ size_t idx_pw(int c, int o) {
  return (((size_t)(c >> 5) * 32 + (o >> 4)) << 9) + swB(c & 31, o & 15);
}

// ---------------- Kernel 1: weight conversion into fragment layouts ----------------
__global__ void cvt_weights(const float* __restrict__ qkv_w, const float* __restrict__ proj_w,
                            _Float16* __restrict__ w16, _Float16* __restrict__ pw) {
  int i = blockIdx.x * blockDim.x + threadIdx.x;
  if (i < 3 * C_ * C_) {
    int o = i >> 9, c = i & 511;
    w16[idx_w(o, c)] = (_Float16)qkv_w[i];
  }
  if (i < C_ * C_) {
    int o = i >> 9, c = i & 511;
    pw[idx_pw(c, o)] = (_Float16)proj_w[i];
  }
}

// ---------------- Kernel 2: GroupNorm -> h (fragment layout) ----------------
__global__ __launch_bounds__(256) void groupnorm_k(const float* __restrict__ x,
                            const float* __restrict__ gw, const float* __restrict__ gb,
                            _Float16* __restrict__ h) {
  const int b = blockIdx.x >> 3;
  const int g = blockIdx.x & 7;
  const size_t base = ((size_t)b * C_ + g * 64) * N_;
  const int cnt = 64 * N_;
  float s = 0.f, q = 0.f;
  for (int i = threadIdx.x; i < cnt; i += 256) {
    float v = x[base + i];
    s += v; q += v * v;
  }
  __shared__ float rs[256], rq[256];
  rs[threadIdx.x] = s; rq[threadIdx.x] = q;
  __syncthreads();
  for (int off = 128; off > 0; off >>= 1) {
    if ((int)threadIdx.x < off) { rs[threadIdx.x] += rs[threadIdx.x + off];
                                  rq[threadIdx.x] += rq[threadIdx.x + off]; }
    __syncthreads();
  }
  const float mean = rs[0] / cnt;
  const float var  = rq[0] / cnt - mean * mean;
  const float inv  = rsqrtf(var + EPSV);
  for (int i = threadIdx.x; i < cnt; i += 256) {
    int c = g * 64 + (i >> 12);
    int n = i & 4095;
    float v = (x[base + i] - mean) * inv * gw[c] + gb[c];
    h[idx_h(b, c, n)] = (_Float16)v;
  }
}

// ---------------- Kernel 3: QKV GEMM (WMMA) -> qT(scaled), k, v ----------------
__global__ __launch_bounds__(128) void qkv_gemm(const _Float16* __restrict__ w16,
        const _Float16* __restrict__ h, const float* __restrict__ qkv_b,
        _Float16* __restrict__ q16, _Float16* __restrict__ k16, _Float16* __restrict__ v16) {
  const int nt = blockIdx.x;                // 64-wide n tile
  const int ot = blockIdx.y;                // 64-tall o tile
  const int b  = blockIdx.z;
  const int w  = threadIdx.x >> 5;
  const int lane = threadIdx.x & 31;
  const int hi = lane >> 4, lo = lane & 15;

  __shared__ v16h sA[2][128];               // 4 A-subtiles x 32 chunks (x2 buffers)
  __shared__ v16h sB[2][128];               // 4 B-subtiles x 32 chunks

  const v16h* wv = (const v16h*)w16;
  const v16h* hv = (const v16h*)h;
  const int tid = threadIdx.x;

  sA[0][tid] = wv[(size_t)(0 * 96 + ot * 4) * 32 + tid];
  sB[0][tid] = hv[((size_t)(b * 16 + 0) * 256 + nt * 4) * 32 + tid];

  v8f acc[4] = {};
  for (int kc = 0; kc < 16; ++kc) {
    __syncthreads();
    if (kc + 1 < 16) {
      sA[(kc + 1) & 1][tid] = wv[(size_t)((kc + 1) * 96 + ot * 4) * 32 + tid];
      sB[(kc + 1) & 1][tid] = hv[((size_t)(b * 16 + kc + 1) * 256 + nt * 4) * 32 + tid];
    }
    v16h aF = sA[kc & 1][w * 32 + lane];
#pragma unroll
    for (int t = 0; t < 4; ++t) {
      v16h bF = sB[kc & 1][t * 32 + lane];
      acc[t] = wmma_f16(aF, bF, acc[t]);
    }
  }
  const int oRow = ot * 64 + w * 16;
#pragma unroll
  for (int t = 0; t < 4; ++t) {
#pragma unroll
    for (int i = 0; i < 8; ++i) {
      int o = oRow + i + 8 * hi;
      int n = nt * 64 + t * 16 + lo;
      float val = acc[t][i] + qkv_b[o];
      if (o < C_) {
        q16[idx_q(b, n, o)] = (_Float16)(val * SCALE);
      } else if (o < 2 * C_) {
        k16[idx_k(b, o - C_, n)] = (_Float16)val;
      } else {
        v16[idx_v(b, n, o - 2 * C_)] = (_Float16)val;
      }
    }
  }
}

// ---------------- Kernel 4: fused flash attention (WMMA + online softmax) ----------------
__global__ __launch_bounds__(128) void attn_k(const _Float16* __restrict__ q16,
        const _Float16* __restrict__ k16, const _Float16* __restrict__ v16,
        _Float16* __restrict__ O16) {
  const int nb = blockIdx.x;                // 16-query-row block, 0..255
  const int b  = blockIdx.y;
  const int w  = threadIdx.x >> 5;          // 4 waves
  const int lane = threadIdx.x & 31;
  const int hi = lane >> 4, lo = lane & 15;
  const int tid = threadIdx.x;

  __shared__ v16h  sQ[512];                 // 16 A-tiles (16KB)
  __shared__ v16h  sK[2][64];               // 4 B-subtiles x2 (4KB)
  __shared__ v16h  sV[1024];                // 32 B-subtiles (32KB)
  __shared__ v16h  sP[64];                  // 2 A-tiles of probs (2KB)
  __shared__ float sS[16 * 64];             // score tile (4KB)
  __shared__ float sMax[16], sSum[16], sFac[16];
  _Float16* sPh = (_Float16*)sP;

  const v16h* qv = (const v16h*)q16;
  const v16h* kv = (const v16h*)k16;
  const v16h* vv = (const v16h*)v16;

  // resident Q strip: 16 chunk-tiles, contiguous in global
  {
    const v16h* src = qv + (size_t)(b * 256 + nb) * 512;
#pragma unroll
    for (int e = 0; e < 4; ++e) sQ[tid + 128 * e] = src[tid + 128 * e];
  }
  if (tid < 16) { sMax[tid] = -1e30f; sSum[tid] = 0.f; }
  __syncthreads();

  v8f accO[8] = {};                         // 16 rows x 128 cols per wave
  for (int mt = 0; mt < 64; ++mt) {
    __builtin_prefetch((const void*)(vv + (size_t)(b * 128 + mt * 2) * 1024), 0, 0);
    // ---- S = Qs^T K over 64-wide key tile (each wave: 16 cols) ----
    if (tid < 64)
      sK[0][tid] = kv[((size_t)(b * 16 + 0) * 256 + mt * 4) * 32 + tid];
    v8f sacc = {};
    for (int kc = 0; kc < 16; ++kc) {
      __syncthreads();
      if (kc + 1 < 16 && tid < 64)
        sK[(kc + 1) & 1][tid] = kv[((size_t)(b * 16 + kc + 1) * 256 + mt * 4) * 32 + tid];
      v16h aF = sQ[kc * 32 + lane];
      v16h bF = sK[kc & 1][w * 32 + lane];
      sacc = wmma_f16(aF, bF, sacc);
    }
#pragma unroll
    for (int i = 0; i < 8; ++i)
      sS[(i + 8 * hi) * 64 + w * 16 + lo] = sacc[i];
    __syncthreads();
    // ---- online softmax: 4 rows per wave ----
#pragma unroll
    for (int rr = 0; rr < 4; ++rr) {
      int r = w * 4 + rr;
      float v0 = sS[r * 64 + lane];
      float v1 = sS[r * 64 + lane + 32];
      float tm = fmaxf(v0, v1);
      for (int off = 16; off > 0; off >>= 1) tm = fmaxf(tm, __shfl_xor(tm, off));
      float om = sMax[r];
      float nm = fmaxf(om, tm);
      float p0 = __expf(v0 - nm), p1 = __expf(v1 - nm);
      sPh[swA(r, lane)]       = (_Float16)p0;   // cols 0..31 -> P-tile 0
      sPh[512 + swA(r, lane)] = (_Float16)p1;   // cols 32..63 -> P-tile 1
      float ps = p0 + p1;
      for (int off = 16; off > 0; off >>= 1) ps += __shfl_xor(ps, off);
      if (lane == 0) {
        float f = __expf(om - nm);
        sSum[r] = sSum[r] * f + ps;
        sMax[r] = nm;
        sFac[r] = f;
      }
    }
    __syncthreads();
    // ---- rescale O accumulators ----
    float fr[8];
#pragma unroll
    for (int i = 0; i < 8; ++i) fr[i] = sFac[i + 8 * hi];
#pragma unroll
    for (int t = 0; t < 8; ++t)
#pragma unroll
      for (int i = 0; i < 8; ++i) accO[t][i] *= fr[i];
    // ---- O += P @ V over two 32-deep chunks ----
    for (int mc = 0; mc < 2; ++mc) {
      const v16h* src = vv + (size_t)(b * 128 + mt * 2 + mc) * 1024;
#pragma unroll
      for (int e = 0; e < 8; ++e) sV[tid + 128 * e] = src[tid + 128 * e];
      __syncthreads();
      v16h aF = sP[mc * 32 + lane];
#pragma unroll
      for (int t = 0; t < 8; ++t) {
        v16h bF = sV[(w * 8 + t) * 32 + lane];
        accO[t] = wmma_f16(aF, bF, accO[t]);
      }
      __syncthreads();
    }
  }
  // ---- normalize, store O in fragment layout ----
  float inv[8];
#pragma unroll
  for (int i = 0; i < 8; ++i) inv[i] = 1.0f / sSum[i + 8 * hi];
#pragma unroll
  for (int t = 0; t < 8; ++t) {
#pragma unroll
    for (int i = 0; i < 8; ++i) {
      int n = nb * 16 + i + 8 * hi;
      int c = w * 128 + t * 16 + lo;
      O16[idx_O(b, n, c)] = (_Float16)(accO[t][i] * inv[i]);
    }
  }
}

// ---------------- Kernel 5: proj GEMM (WMMA) + bias + residual ----------------
__global__ __launch_bounds__(128) void proj_gemm(const _Float16* __restrict__ O16,
        const _Float16* __restrict__ pw, const float* __restrict__ proj_b,
        const float* __restrict__ x, float* __restrict__ out) {
  const int nt = blockIdx.x;                // n tile
  const int ot = blockIdx.y;                // o tile
  const int b  = blockIdx.z;
  const int w  = threadIdx.x >> 5;
  const int lane = threadIdx.x & 31;
  const int hi = lane >> 4, lo = lane & 15;
  const int tid = threadIdx.x;

  __shared__ v16h sA[2][128];
  __shared__ v16h sB[2][128];

  const v16h* Ov = (const v16h*)O16;
  const v16h* pv = (const v16h*)pw;

  {
    int s = tid >> 5, off = tid & 31;
    sA[0][tid] = Ov[((size_t)(b * 256 + nt * 4 + s) * 16 + 0) * 32 + off];
    sB[0][tid] = pv[(size_t)(0 * 32 + ot * 4) * 32 + tid];
  }
  v8f acc[4] = {};
  for (int kc = 0; kc < 16; ++kc) {
    __syncthreads();
    if (kc + 1 < 16) {
      int s = tid >> 5, off = tid & 31;
      sA[(kc + 1) & 1][tid] = Ov[((size_t)(b * 256 + nt * 4 + s) * 16 + kc + 1) * 32 + off];
      sB[(kc + 1) & 1][tid] = pv[(size_t)((kc + 1) * 32 + ot * 4) * 32 + tid];
    }
    v16h aF = sA[kc & 1][w * 32 + lane];
#pragma unroll
    for (int t = 0; t < 4; ++t) {
      v16h bF = sB[kc & 1][t * 32 + lane];
      acc[t] = wmma_f16(aF, bF, acc[t]);
    }
  }
#pragma unroll
  for (int t = 0; t < 4; ++t) {
#pragma unroll
    for (int i = 0; i < 8; ++i) {
      int n = nt * 64 + w * 16 + i + 8 * hi;
      int o = ot * 64 + t * 16 + lo;
      size_t idx = ((size_t)b * C_ + o) * N_ + n;
      out[idx] = x[idx] + acc[t][i] + proj_b[o];
    }
  }
}

// ---------------- launch ----------------
extern "C" void kernel_launch(void* const* d_in, const int* in_sizes, int n_in,
                              void* d_out, int out_size, void* d_ws, size_t ws_size,
                              hipStream_t stream) {
  (void)in_sizes; (void)n_in; (void)out_size; (void)ws_size;
  const float* x      = (const float*)d_in[0];
  const float* gn_w   = (const float*)d_in[1];
  const float* gn_b   = (const float*)d_in[2];
  const float* qkv_w  = (const float*)d_in[3];
  const float* qkv_b  = (const float*)d_in[4];
  const float* proj_w = (const float*)d_in[5];
  const float* proj_b = (const float*)d_in[6];
  float* out = (float*)d_out;

  _Float16* ws  = (_Float16*)d_ws;
  _Float16* h   = ws;                       // fragment-tiled layouts (see idx_*)
  _Float16* q16 = ws + SEG;
  _Float16* k16 = ws + 2 * SEG;
  _Float16* v16 = ws + 3 * SEG;
  _Float16* O16 = ws + 4 * SEG;
  _Float16* w16 = ws + 5 * SEG;
  _Float16* pw  = w16 + (size_t)3 * C_ * C_;

  cvt_weights<<<dim3((3 * C_ * C_ + 255) / 256), dim3(256), 0, stream>>>(qkv_w, proj_w, w16, pw);
  groupnorm_k<<<dim3(B_ * 8), dim3(256), 0, stream>>>(x, gn_w, gn_b, h);
  qkv_gemm<<<dim3(N_ / 64, 3 * C_ / 64, B_), dim3(128), 0, stream>>>(w16, h, qkv_b, q16, k16, v16);
  attn_k<<<dim3(N_ / 16, B_), dim3(128), 0, stream>>>(q16, k16, v16, O16);
  proj_gemm<<<dim3(N_ / 64, C_ / 64, B_), dim3(128), 0, stream>>>(O16, pw, proj_b, x, out);
}
